// Attend_28982439313688
// MI455X (gfx1250) — matrix operand order repaired and checked by reading, not compiled, not run on hardware
//
#include <hip/hip_runtime.h>
#include <hip/hip_bf16.h>
#include <math.h>

// L2-distance flash attention for gfx1250 (MI455X).
// B=2 H=16 N=2048 D=64 fp32 in/out; f16 WMMA with fp32 accumulation.
// Two-phase: (1) K/V -> f16 (V transposed) + |k|^2+mask table in workspace,
// (2) double-buffered async-to-LDS tile pipeline feeding v_wmma_f32_16x16x32_f16.

typedef _Float16 v16h __attribute__((ext_vector_type(16)));
typedef _Float16 v8h  __attribute__((ext_vector_type(8)));
typedef _Float16 v4h  __attribute__((ext_vector_type(4)));
typedef float    v8f  __attribute__((ext_vector_type(8)));
typedef int      i4v  __attribute__((vector_size(16)));   // matches builtin proto

#define BB   2
#define HH   16
#define NN   2048
#define DD   64
#define BM   128        // query rows per block (8 waves x 16)
#define BN   64         // keys per tile (4 WMMA subtiles)
#define NWAVE 8
#define LDK  72         // padded LDS row stride in halves (144 B = 9*16 B)
#define T_TILES (NN / BN)

union F8  { v8f  v; float    f[8];  };
union H16 { v16h v; v8h      h[2]; _Float16 e[16]; };

// ---- async copy (global -> LDS), 16 B per lane, guarded ----------------------
#if __has_builtin(__builtin_amdgcn_global_load_async_to_lds_b128)
#define HAVE_ASYNC 1
#define ASYNC_CP16(g, l)                                                      \
    __builtin_amdgcn_global_load_async_to_lds_b128(                           \
        (__attribute__((address_space(1))) i4v*)(g),                          \
        (__attribute__((address_space(3))) i4v*)(l), 0, 0)
#else
#define HAVE_ASYNC 0
#define ASYNC_CP16(g, l)                                                      \
    do { uint4 _t = *(const uint4*)(g); *(uint4*)(l) = _t; } while (0)
#endif

#if HAVE_ASYNC
#if __has_builtin(__builtin_amdgcn_s_wait_asynccnt)
#define WAIT_ASYNC(n) __builtin_amdgcn_s_wait_asynccnt(n)
#else
#define WAIT_ASYNC(n) asm volatile("s_wait_asynccnt %0" :: "i"(n) : "memory")
#endif
#else
#define WAIT_ASYNC(n) do { } while (0)
#endif

// =============================================================================
// Phase 1: K -> f16 [bh][n][d], V -> f16 transposed [bh][d][n], |k|^2 + mask.
// =============================================================================
__global__ __launch_bounds__(256, 1)
void preprocess_kv(const float* __restrict__ k, const float* __restrict__ v,
                   const unsigned char* __restrict__ mask,
                   _Float16* __restrict__ kf, _Float16* __restrict__ vtf,
                   float* __restrict__ ksq)
{
    __shared__ float    ksqt[BN];
    __shared__ _Float16 vstage[DD][LDK];      // V^T staging for coalesced writes

    const int tid = threadIdx.x;
    const int bh  = blockIdx.y;
    const int b   = bh / HH;
    const int j0  = blockIdx.x * BN;

    const float* khd = k + (size_t)bh * NN * DD;
    const float* vhd = v + (size_t)bh * NN * DD;

    if (tid < BN) ksqt[tid] = 0.f;
    __syncthreads();

    #pragma unroll
    for (int t = 0; t < 4; ++t) {
        int i   = tid + t * 256;              // float4 index over 64x16
        int key = i >> 4;
        int d4  = (i & 15) * 4;
        float4 kx = *(const float4*)(khd + (size_t)(j0 + key) * DD + d4);
        v4h k4 = { (_Float16)kx.x, (_Float16)kx.y, (_Float16)kx.z, (_Float16)kx.w };
        *(v4h*)(kf + ((size_t)bh * NN + j0 + key) * DD + d4) = k4;
        atomicAdd(&ksqt[key], kx.x * kx.x + kx.y * kx.y + kx.z * kx.z + kx.w * kx.w);

        float4 vx = *(const float4*)(vhd + (size_t)(j0 + key) * DD + d4);
        vstage[d4 + 0][key] = (_Float16)vx.x;
        vstage[d4 + 1][key] = (_Float16)vx.y;
        vstage[d4 + 2][key] = (_Float16)vx.z;
        vstage[d4 + 3][key] = (_Float16)vx.w;
    }
    __syncthreads();

    // coalesced V^T writes: rows are d, 64 keys (128 B) each
    #pragma unroll
    for (int t = 0; t < 2; ++t) {
        int idx = tid + t * 256;              // 0..511
        int d   = idx >> 3;
        int seg = (idx & 7) * 8;
        *(v8h*)(vtf + ((size_t)bh * DD + d) * NN + j0 + seg) =
            *(const v8h*)&vstage[d][seg];
    }
    if (tid < BN) {
        int j = j0 + tid;
        ksq[(size_t)bh * NN + j] = mask[(size_t)b * NN + j] ? 3.0e34f : ksqt[tid];
    }
}

// =============================================================================
// Phase 2: flash attention main loop (double-buffered async LDS tiles).
// =============================================================================
__global__ __launch_bounds__(256, 1)
void attend_l2_fast(const float* __restrict__ q,
                    const _Float16* __restrict__ kf,
                    const _Float16* __restrict__ vtf,
                    const float* __restrict__ ksqg,
                    float* __restrict__ out)
{
    __shared__ _Float16 kt[2][BN][LDK];          // 18 KB  K tiles   [key][d]
    __shared__ _Float16 vt[2][DD][LDK];          // 18 KB  V^T tiles [d][key]
    __shared__ float    ksqa[NN];                //  8 KB
    __shared__ _Float16 pl[NWAVE][16][LDK];      // 18 KB  P staging

    const int tid  = threadIdx.x;
    const int lane = tid & 31;
    const int wave = tid >> 5;
    const int hi   = lane >> 4;
    const int ln   = lane & 15;

    const int bh    = blockIdx.y;
    const int qbase = blockIdx.x * BM;

    const float*    qh = q   + (size_t)bh * NN * DD;
    const _Float16* kh = kf  + (size_t)bh * NN * DD;
    const _Float16* vh = vtf + (size_t)bh * DD * NN;
    float*          oh = out + (size_t)bh * NN * DD;

    // ---- issue tile 0 copies ASAP (4 async ops per thread per tile) ----
    auto issue_tile = [&](int jt, int buf) {
        #pragma unroll
        for (int t = 0; t < 2; ++t) {
            int o   = tid + t * 256;             // 0..511
            int row = o >> 3;                    // 64 rows, 128 B each
            int seg = (o & 7) * 8;               // 8 halves = 16 B
            ASYNC_CP16(kh + (size_t)(jt * BN + row) * DD + seg, &kt[buf][row][seg]);
            ASYNC_CP16(vh + (size_t)row * NN + jt * BN + seg,   &vt[buf][row][seg]);
        }
    };
    issue_tile(0, 0);

    // ---- stage |k|^2 table ----
    #pragma unroll
    for (int t = 0; t < NN / 256; ++t) {
        int j = tid + t * 256;
        ksqa[j] = ksqg[(size_t)bh * NN + j];
    }

    // ---- Q A-fragments (16x32 f16 layout) + |q|^2 ----
    const int qrow = qbase + wave * 16 + ln;
    const float* qr = qh + (size_t)qrow * DD;
    const int off = hi * 8;

    H16 afr[2];
    float qs_part = 0.f;
    #pragma unroll
    for (int s = 0; s < 4; ++s) {
        int kb = off + s * 16;
        const float4* p4 = (const float4*)(qr + kb);
        float4 x0 = p4[0], x1 = p4[1];
        qs_part += x0.x * x0.x + x0.y * x0.y + x0.z * x0.z + x0.w * x0.w
                 + x1.x * x1.x + x1.y * x1.y + x1.z * x1.z + x1.w * x1.w;
        int fi = s >> 1, hb = (s & 1) * 8;
        afr[fi].e[hb + 0] = (_Float16)x0.x;  afr[fi].e[hb + 1] = (_Float16)x0.y;
        afr[fi].e[hb + 2] = (_Float16)x0.z;  afr[fi].e[hb + 3] = (_Float16)x0.w;
        afr[fi].e[hb + 4] = (_Float16)x1.x;  afr[fi].e[hb + 5] = (_Float16)x1.y;
        afr[fi].e[hb + 6] = (_Float16)x1.z;  afr[fi].e[hb + 7] = (_Float16)x1.w;
    }
    float qsq = qs_part + __shfl_xor(qs_part, 16, 32);
    float qsqv[8];
    #pragma unroll
    for (int r = 0; r < 8; ++r) qsqv[r] = __shfl(qsq, r + hi * 8, 32);

    // ---- running softmax state + accumulators ----
    F8 o[4];
    float m[8], l[8];
    #pragma unroll
    for (int r = 0; r < 8; ++r) { m[r] = -3.0e38f; l[r] = 0.f; }
    #pragma unroll
    for (int dt = 0; dt < 4; ++dt)
        #pragma unroll
        for (int r = 0; r < 8; ++r) o[dt].f[r] = 0.f;

    const float sc2 = 2.0f * 0.125f;             // 2 * D^-0.5

    for (int jt = 0; jt < T_TILES; ++jt) {
        const int cur = jt & 1;
        const int j0  = jt * BN;

        if (jt + 1 < T_TILES) {
            if (jt) __syncthreads();             // readers of buf cur^1 done
            issue_tile(jt + 1, cur ^ 1);
            WAIT_ASYNC(4);                       // current tile's 4 ops landed
        } else {
            WAIT_ASYNC(0);
        }
        __syncthreads();                         // whole block's tile in LDS

        if (jt + 2 < T_TILES)                    // warm L2 for tile jt+2
            __builtin_prefetch((const char*)(kh + (size_t)(j0 + 2 * BN) * DD) + tid * 16, 0, 1);

        // ---- S = 2*scale*(Q K^T) - |q|^2 - |k|^2 ----
        F8 sfr[4];
        #pragma unroll
        for (int t = 0; t < 4; ++t) {
            int key = t * 16 + ln;
            int lo  = hi * 8;
            H16 b0, b1;
            b0.h[0] = *(const v8h*)&kt[cur][key][lo];
            b0.h[1] = *(const v8h*)&kt[cur][key][lo + 16];
            b1.h[0] = *(const v8h*)&kt[cur][key][lo + 32];
            b1.h[1] = *(const v8h*)&kt[cur][key][lo + 48];
            v8f c = {};
            c = __builtin_amdgcn_wmma_f32_16x16x32_f16(false, afr[0].v, false, b0.v,
                                                       (short)0, c, false, false);
            c = __builtin_amdgcn_wmma_f32_16x16x32_f16(false, afr[1].v, false, b1.v,
                                                       (short)0, c, false, false);
            float ksub = ksqa[j0 + key];
            F8 sf; sf.v = c;
            #pragma unroll
            for (int r = 0; r < 8; ++r)
                sf.f[r] = sc2 * sf.f[r] - qsqv[r] - ksub;
            sfr[t] = sf;
        }

        // ---- online softmax ----
        float alpha[8], psum[8];
        #pragma unroll
        for (int r = 0; r < 8; ++r) {
            float x = fmaxf(fmaxf(sfr[0].f[r], sfr[1].f[r]),
                            fmaxf(sfr[2].f[r], sfr[3].f[r]));
            x = fmaxf(x, __shfl_xor(x, 1, 32));
            x = fmaxf(x, __shfl_xor(x, 2, 32));
            x = fmaxf(x, __shfl_xor(x, 4, 32));
            x = fmaxf(x, __shfl_xor(x, 8, 32));
            float nm = fmaxf(m[r], x);
            alpha[r] = __expf(m[r] - nm);
            m[r] = nm;
            psum[r] = 0.f;
        }
        #pragma unroll
        for (int t = 0; t < 4; ++t)
            #pragma unroll
            for (int r = 0; r < 8; ++r) {
                float p = __expf(sfr[t].f[r] - m[r]);
                sfr[t].f[r] = p;
                psum[r] += p;
            }
        #pragma unroll
        for (int r = 0; r < 8; ++r) {
            float x = psum[r];
            x += __shfl_xor(x, 1, 32);
            x += __shfl_xor(x, 2, 32);
            x += __shfl_xor(x, 4, 32);
            x += __shfl_xor(x, 8, 32);
            l[r] = l[r] * alpha[r] + x;
        }
        #pragma unroll
        for (int dt = 0; dt < 4; ++dt)
            #pragma unroll
            for (int r = 0; r < 8; ++r) o[dt].f[r] *= alpha[r];

        // ---- P: C/D layout -> A layout through per-wave LDS ----
        #pragma unroll
        for (int t = 0; t < 4; ++t)
            #pragma unroll
            for (int r = 0; r < 8; ++r)
                pl[wave][r + hi * 8][t * 16 + ln] = (_Float16)sfr[t].f[r];

        H16 pa0, pa1;
        #pragma unroll
        for (int s = 0; s < 4; ++s) {
            int kb = off + s * 16;
            v8h seg = *(const v8h*)&pl[wave][ln][kb];
            (s >> 1 ? pa1 : pa0).h[s & 1] = seg;
        }

        // ---- O += P V ----
        #pragma unroll
        for (int dt = 0; dt < 4; ++dt) {
            int dcol = dt * 16 + ln;
            int lo   = hi * 8;
            H16 vb0, vb1;
            vb0.h[0] = *(const v8h*)&vt[cur][dcol][lo];
            vb0.h[1] = *(const v8h*)&vt[cur][dcol][lo + 16];
            vb1.h[0] = *(const v8h*)&vt[cur][dcol][lo + 32];
            vb1.h[1] = *(const v8h*)&vt[cur][dcol][lo + 48];
            v8f c = o[dt].v;
            c = __builtin_amdgcn_wmma_f32_16x16x32_f16(false, pa0.v, false, vb0.v,
                                                       (short)0, c, false, false);
            c = __builtin_amdgcn_wmma_f32_16x16x32_f16(false, pa1.v, false, vb1.v,
                                                       (short)0, c, false, false);
            o[dt].v = c;
        }
    }

    // ---- finalize ----
    float inv[8];
    #pragma unroll
    for (int r = 0; r < 8; ++r) inv[r] = 1.0f / l[r];
    #pragma unroll
    for (int dt = 0; dt < 4; ++dt)
        #pragma unroll
        for (int r = 0; r < 8; ++r) {
            int row = qbase + wave * 16 + r + hi * 8;
            oh[(size_t)row * DD + dt * 16 + ln] = o[dt].f[r] * inv[r];
        }
}

// =============================================================================
// Fallback: round-1 single-kernel version (used when ws is too small).
// =============================================================================
__global__ __launch_bounds__(256, 1)
void attend_l2_flash_kernel(const float* __restrict__ q,
                            const float* __restrict__ k,
                            const float* __restrict__ v,
                            const unsigned char* __restrict__ mask,
                            float* __restrict__ out)
{
    __shared__ float    ksqa[NN];
    __shared__ _Float16 kt[BN][LDK];
    __shared__ _Float16 vt[DD][LDK];
    __shared__ _Float16 pl[NWAVE][16][LDK];

    const int tid  = threadIdx.x;
    const int lane = tid & 31;
    const int wave = tid >> 5;
    const int hi   = lane >> 4;
    const int ln   = lane & 15;

    const int bh    = blockIdx.y;
    const int b     = bh / HH;
    const int qbase = blockIdx.x * BM;

    const float* qh = q + (size_t)bh * NN * DD;
    const float* kh = k + (size_t)bh * NN * DD;
    const float* vh = v + (size_t)bh * NN * DD;
    float*       oh = out + (size_t)bh * NN * DD;
    const unsigned char* mb = mask + (size_t)b * NN;

    #pragma unroll
    for (int t = 0; t < NN / 256; ++t) {
        int j = tid + t * 256;
        const float4* kr = (const float4*)(kh + (size_t)j * DD);
        float s = 0.f;
        #pragma unroll
        for (int c = 0; c < DD / 4; ++c) {
            float4 x = kr[c];
            s += x.x * x.x + x.y * x.y + x.z * x.z + x.w * x.w;
        }
        ksqa[j] = mb[j] ? 3.0e34f : s;
    }

    const int qrow = qbase + wave * 16 + ln;
    const float* qr = qh + (size_t)qrow * DD;
    const int off = hi * 8;

    H16 afr[2];
    float qs_part = 0.f;
    #pragma unroll
    for (int s = 0; s < 4; ++s) {
        int kb = off + s * 16;
        const float4* p4 = (const float4*)(qr + kb);
        float4 x0 = p4[0], x1 = p4[1];
        qs_part += x0.x * x0.x + x0.y * x0.y + x0.z * x0.z + x0.w * x0.w
                 + x1.x * x1.x + x1.y * x1.y + x1.z * x1.z + x1.w * x1.w;
        int fi = s >> 1, hb = (s & 1) * 8;
        afr[fi].e[hb + 0] = (_Float16)x0.x;  afr[fi].e[hb + 1] = (_Float16)x0.y;
        afr[fi].e[hb + 2] = (_Float16)x0.z;  afr[fi].e[hb + 3] = (_Float16)x0.w;
        afr[fi].e[hb + 4] = (_Float16)x1.x;  afr[fi].e[hb + 5] = (_Float16)x1.y;
        afr[fi].e[hb + 6] = (_Float16)x1.z;  afr[fi].e[hb + 7] = (_Float16)x1.w;
    }
    float qsq = qs_part + __shfl_xor(qs_part, 16, 32);
    float qsqv[8];
    #pragma unroll
    for (int r = 0; r < 8; ++r) qsqv[r] = __shfl(qsq, r + hi * 8, 32);

    F8 o[4];
    float m[8], l[8];
    #pragma unroll
    for (int r = 0; r < 8; ++r) { m[r] = -3.0e38f; l[r] = 0.f; }
    #pragma unroll
    for (int dt = 0; dt < 4; ++dt)
        #pragma unroll
        for (int r = 0; r < 8; ++r) o[dt].f[r] = 0.f;

    const float sc2 = 2.0f * 0.125f;

    for (int jt = 0; jt < T_TILES; ++jt) {
        const int j0 = jt * BN;
        __syncthreads();
        #pragma unroll
        for (int t = 0; t < 4; ++t) {
            int i   = tid + t * 256;
            int key = i >> 4;
            int d4  = (i & 15) * 4;
            float4 kx = *(const float4*)(kh + (size_t)(j0 + key) * DD + d4);
            v4h k4 = { (_Float16)kx.x, (_Float16)kx.y, (_Float16)kx.z, (_Float16)kx.w };
            *(v4h*)&kt[key][d4] = k4;
            float4 vx = *(const float4*)(vh + (size_t)(j0 + key) * DD + d4);
            vt[d4 + 0][key] = (_Float16)vx.x;
            vt[d4 + 1][key] = (_Float16)vx.y;
            vt[d4 + 2][key] = (_Float16)vx.z;
            vt[d4 + 3][key] = (_Float16)vx.w;
        }
        __syncthreads();

        F8 sfr[4];
        #pragma unroll
        for (int t = 0; t < 4; ++t) {
            int key = t * 16 + ln;
            int lo  = hi * 8;
            H16 b0, b1;
            b0.h[0] = *(const v8h*)&kt[key][lo];
            b0.h[1] = *(const v8h*)&kt[key][lo + 16];
            b1.h[0] = *(const v8h*)&kt[key][lo + 32];
            b1.h[1] = *(const v8h*)&kt[key][lo + 48];
            v8f c = {};
            c = __builtin_amdgcn_wmma_f32_16x16x32_f16(false, afr[0].v, false, b0.v,
                                                       (short)0, c, false, false);
            c = __builtin_amdgcn_wmma_f32_16x16x32_f16(false, afr[1].v, false, b1.v,
                                                       (short)0, c, false, false);
            float ksub = ksqa[j0 + key];
            F8 sf; sf.v = c;
            #pragma unroll
            for (int r = 0; r < 8; ++r)
                sf.f[r] = sc2 * sf.f[r] - qsqv[r] - ksub;
            sfr[t] = sf;
        }

        float alpha[8], psum[8];
        #pragma unroll
        for (int r = 0; r < 8; ++r) {
            float x = fmaxf(fmaxf(sfr[0].f[r], sfr[1].f[r]),
                            fmaxf(sfr[2].f[r], sfr[3].f[r]));
            x = fmaxf(x, __shfl_xor(x, 1, 32));
            x = fmaxf(x, __shfl_xor(x, 2, 32));
            x = fmaxf(x, __shfl_xor(x, 4, 32));
            x = fmaxf(x, __shfl_xor(x, 8, 32));
            float nm = fmaxf(m[r], x);
            alpha[r] = __expf(m[r] - nm);
            m[r] = nm;
            psum[r] = 0.f;
        }
        #pragma unroll
        for (int t = 0; t < 4; ++t)
            #pragma unroll
            for (int r = 0; r < 8; ++r) {
                float p = __expf(sfr[t].f[r] - m[r]);
                sfr[t].f[r] = p;
                psum[r] += p;
            }
        #pragma unroll
        for (int r = 0; r < 8; ++r) {
            float x = psum[r];
            x += __shfl_xor(x, 1, 32);
            x += __shfl_xor(x, 2, 32);
            x += __shfl_xor(x, 4, 32);
            x += __shfl_xor(x, 8, 32);
            l[r] = l[r] * alpha[r] + x;
        }
        #pragma unroll
        for (int dt = 0; dt < 4; ++dt)
            #pragma unroll
            for (int r = 0; r < 8; ++r) o[dt].f[r] *= alpha[r];

        #pragma unroll
        for (int t = 0; t < 4; ++t)
            #pragma unroll
            for (int r = 0; r < 8; ++r)
                pl[wave][r + hi * 8][t * 16 + ln] = (_Float16)sfr[t].f[r];

        H16 pa0, pa1;
        #pragma unroll
        for (int s = 0; s < 4; ++s) {
            int kb = off + s * 16;
            v8h seg = *(const v8h*)&pl[wave][ln][kb];
            (s >> 1 ? pa1 : pa0).h[s & 1] = seg;
        }

        #pragma unroll
        for (int dt = 0; dt < 4; ++dt) {
            int dcol = dt * 16 + ln;
            int lo   = hi * 8;
            H16 vb0, vb1;
            vb0.h[0] = *(const v8h*)&vt[dcol][lo];
            vb0.h[1] = *(const v8h*)&vt[dcol][lo + 16];
            vb1.h[0] = *(const v8h*)&vt[dcol][lo + 32];
            vb1.h[1] = *(const v8h*)&vt[dcol][lo + 48];
            v8f c = o[dt].v;
            c = __builtin_amdgcn_wmma_f32_16x16x32_f16(false, pa0.v, false, vb0.v,
                                                       (short)0, c, false, false);
            c = __builtin_amdgcn_wmma_f32_16x16x32_f16(false, pa1.v, false, vb1.v,
                                                       (short)0, c, false, false);
            o[dt].v = c;
        }
    }

    float inv[8];
    #pragma unroll
    for (int r = 0; r < 8; ++r) inv[r] = 1.0f / l[r];
    #pragma unroll
    for (int dt = 0; dt < 4; ++dt)
        #pragma unroll
        for (int r = 0; r < 8; ++r) {
            int row = qbase + wave * 16 + r + hi * 8;
            oh[(size_t)row * DD + dt * 16 + ln] = o[dt].f[r] * inv[r];
        }
}

// =============================================================================
extern "C" void kernel_launch(void* const* d_in, const int* in_sizes, int n_in,
                              void* d_out, int out_size, void* d_ws, size_t ws_size,
                              hipStream_t stream) {
    (void)in_sizes; (void)n_in; (void)out_size;
    const float* q = (const float*)d_in[0];
    const float* k = (const float*)d_in[1];
    const float* v = (const float*)d_in[2];
    const unsigned char* mask = (const unsigned char*)d_in[3];  // jax bool, 1 byte
    float* out = (float*)d_out;

    const size_t szH   = (size_t)BB * HH * NN * DD * sizeof(_Float16);  // 8 MB
    const size_t szKsq = (size_t)BB * HH * NN * sizeof(float);          // 256 KB
    const size_t need  = 2 * szH + szKsq;

    if (ws_size >= need) {
        _Float16* kf  = (_Float16*)d_ws;
        _Float16* vtf = (_Float16*)((char*)d_ws + szH);
        float*    ksq = (float*)((char*)d_ws + 2 * szH);
        dim3 gpre(NN / BN, BB * HH);
        preprocess_kv<<<gpre, 256, 0, stream>>>(k, v, mask, kf, vtf, ksq);
        dim3 grid(NN / BM, BB * HH);
        attend_l2_fast<<<grid, 256, 0, stream>>>(q, kf, vtf, ksq, out);
    } else {
        dim3 grid(NN / BM, BB * HH);
        attend_l2_flash_kernel<<<grid, 256, 0, stream>>>(q, k, v, mask, out);
    }
}
